// QueryOnlyTTT_62328565400272
// MI455X (gfx1250) — matrix operand order repaired and checked by reading, not compiled, not run on hardware
//
#include <hip/hip_runtime.h>

// ---------- types ----------
typedef __bf16 bf16;
typedef bf16  v4bf  __attribute__((ext_vector_type(4)));
typedef bf16  v8bf  __attribute__((ext_vector_type(8)));
typedef bf16  v16bf __attribute__((ext_vector_type(16)));
typedef float v8f   __attribute__((ext_vector_type(8)));
typedef unsigned int v4u __attribute__((ext_vector_type(4)));
typedef int  v4i __attribute__((ext_vector_type(4)));
typedef int  v8i __attribute__((ext_vector_type(8)));

#define DEV __device__ __forceinline__

// ---------- problem constants (from reference setup) ----------
constexpr int   Bb     = 2;
constexpr int   Hh     = 32;
constexpr int   BH     = Bb * Hh;          // 64
constexpr int   TQ     = 128;              // span length k
constexpr int   DH     = 64;               // head dim
constexpr int   Dd     = 2048;             // hidden dim
constexpr long  TKV    = 8192;             // frozen KV length
constexpr int   SPLIT  = 4;                // kv splits (flash-decoding)
constexpr int   TKS    = (int)(TKV / SPLIT); // 2048 kv per split
constexpr int   TT     = 128;              // kv tile
constexpr int   NTS    = TKS / TT;         // 16 tiles per split
constexpr int   NSTEPS = 16;
constexpr float SCALE  = 0.125f;           // 1/sqrt(64)
constexpr float LRc    = 0.005f;
constexpr float INVN   = 1.0f / (float)(BH * TQ * DH); // 1/524288

// ---------- small helpers ----------
DEV v8f vzero8() {
  v8f v;
#pragma unroll
  for (int i = 0; i < 8; ++i) v[i] = 0.f;
  return v;
}

DEV v8f wmma_bf16(v16bf a, v16bf b, v8f c) {
  return __builtin_amdgcn_wmma_f32_16x16x32_bf16(
      false, a, false, b, (short)0, c, false, false);
}

// DPP row_ror butterfly reductions across the 16-lane half (no LDS traffic)
#define DPP_ROR(v, n) \
  __int_as_float(__builtin_amdgcn_update_dpp(0, __float_as_int(v), 0x120 + (n), 0xF, 0xF, true))

DEV float redsum16(float v) {
  v += DPP_ROR(v, 1);
  v += DPP_ROR(v, 2);
  v += DPP_ROR(v, 4);
  v += DPP_ROR(v, 8);
  return v;
}
DEV float redmax16(float v) {
  v = fmaxf(v, DPP_ROR(v, 1));
  v = fmaxf(v, DPP_ROR(v, 2));
  v = fmaxf(v, DPP_ROR(v, 4));
  v = fmaxf(v, DPP_ROR(v, 8));
  return v;
}

// A-matrix (16x32, 16-bit) lane layout (ISA 7.12.2)
DEV int amap(int j, int half) {
  return j < 8 ? (half ? j + 8 : j) : (half ? j + 16 : j + 8);
}

// B-matrix (32x16) fragment: element j -> k = amap(j, half), n = lane%16.
DEV v16bf bfrag(const bf16* mat, int ld, int kbase, int nbase, int lane) {
  const int half = (lane >> 4) & 1, n = lane & 15;
  const bf16* p = mat + (nbase + n) * ld + kbase;
  union { v16bf v; v8bf h[2]; } u;
  u.h[0] = *(const v8bf*)(p + half * 8);
  u.h[1] = *(const v8bf*)(p + 16 + half * 8);
  return u.v;
}

// A-matrix fragment from an LDS strip stored [16][ld] row-major.
DEV v16bf afrag(const bf16* strip, int ld, int kbase, int lane) {
  const int half = (lane >> 4) & 1, r = lane & 15;
  const bf16* p = strip + r * ld + kbase;
  union { v16bf v; v8bf h[2]; } u;
  u.h[0] = *(const v8bf*)(p + (half ? 8  : 0));
  u.h[1] = *(const v8bf*)(p + (half ? 24 : 16));
  return u.v;
}

// ---------- Tensor Data Mover: 2-D tile (128 rows x 64 bf16) -> LDS ----------
// D# per ISA ch.8: group0 = {count/flags, lds_addr, global_addr, type},
// group1 = {mask+data_size, dims/strides}; groups 2-3 zero (<=2-D tensor).
DEV unsigned lds_byte_off(const void* p) {
  // flat LDS aperture: addr[31:0] is the LDS byte address (ISA 10.2)
  return (unsigned)(uintptr_t)p;
}

DEV void tdm_load_tile(const bf16* gaddr, unsigned ldsOff) {
  unsigned long long ga = (unsigned long long)(uintptr_t)gaddr;
  v4u g0;
  g0[0] = 1u;                                    // count=1 (valid user D#)
  g0[1] = ldsOff;                                // lds_addr (bytes)
  g0[2] = (unsigned)(ga & 0xFFFFFFFFu);          // global_addr[31:0]
  g0[3] = (unsigned)((ga >> 32) & 0x01FFFFFFu)   // global_addr[56:32]
        | (2u << 30);                            // type = 2 ("image")
  v8i g1;
  g1[0] = 0x00010000;          // wg_mask=0, data_size=1 (2 bytes)
  g1[1] = (int)(64u << 16);    // tensor_dim0 = 64 (lo16)
  g1[2] = (int)(128u << 16);   // tensor_dim0 hi=0 | tensor_dim1 lo = 128
  g1[3] = (int)(64u << 16);    // tensor_dim1 hi=0 | tile_dim0 = 64
  g1[4] = 128;                 // tile_dim1 = 128, tile_dim2 = 0
  g1[5] = 64;                  // tensor_dim0_stride = 64 (lo32)
  g1[6] = 0;                   // dim0_stride hi | dim1_stride lo
  g1[7] = 0;                   // dim1_stride hi
  v4i z4 = {0, 0, 0, 0};
  v8i z8 = {0, 0, 0, 0, 0, 0, 0, 0};
  // this toolchain declares the 6-arg form:
  // (uint32x4 g0, int32x8 g1, int32x4, int32x4, int32x8, i32 cpol)
  __builtin_amdgcn_tensor_load_to_lds(g0, g1, z4, z4, z8, 0);
}

// ---------- VALU staging for the transposed layouts (TDM cannot transpose) --
DEV void stage_vtr(const bf16* Vg, int t0, bf16* Vtr, int tid) {
#pragma unroll
  for (int it = 0; it < 8; ++it) {
    const int e  = it * 256 + tid;
    const int t  = e >> 4;
    const int d4 = (e & 15) * 4;
    v4bf vq = *(const v4bf*)(Vg + (size_t)(t0 + t) * DH + d4);
#pragma unroll
    for (int j = 0; j < 4; ++j) Vtr[(d4 + j) * TT + t] = vq[j];
  }
}
DEV void stage_ktr(const bf16* Kg, int t0, bf16* Ktr, int tid) {
#pragma unroll
  for (int it = 0; it < 8; ++it) {
    const int e  = it * 256 + tid;
    const int t  = e >> 4;
    const int d4 = (e & 15) * 4;
    v4bf kq = *(const v4bf*)(Kg + (size_t)(t0 + t) * DH + d4);
#pragma unroll
    for (int j = 0; j < 4; ++j) Ktr[(d4 + j) * TT + t] = kq[j];
  }
}

// ============ forward partial: flash attention over one kv split ============
__global__ __launch_bounds__(256, 1)
void fwd_partial_kernel(const float* __restrict__ qbuf,
                        const bf16* __restrict__ Kbf,
                        const bf16* __restrict__ Vbf,
                        float* __restrict__ m_part,
                        float* __restrict__ l_part,
                        float* __restrict__ O_part) {
  __shared__ alignas(32) bf16 KrowBuf[2][TT * DH];    // TDM double buffer
  __shared__ alignas(32) bf16 Vtr [DH * TT];
  __shared__ alignas(32) bf16 scratch[8 * 16 * 128];  // per-wave P strip

  const int blk = blockIdx.x;               // bh*SPLIT + sp
  const int bh = blk >> 2, sp = blk & 3;
  const int tid = threadIdx.x;
  const int wave = tid >> 5, lane = tid & 31;
  const int half = lane >> 4, ncol = lane & 15;
  const int r0 = wave * 16;

  const float* qb = qbuf + (size_t)bh * TQ * DH;
  const bf16* Kg  = Kbf + ((size_t)bh * TKV + (size_t)sp * TKS) * DH;
  const bf16* Vg  = Vbf + ((size_t)bh * TKV + (size_t)sp * TKS) * DH;
  bf16* myscr     = scratch + wave * (16 * 128);

  // q A-fragments (f32 global -> bf16, once)
  v16bf aq[2];
  {
    const int r = r0 + (lane & 15);
#pragma unroll
    for (int c = 0; c < 2; ++c)
#pragma unroll
      for (int j = 0; j < 16; ++j)
        aq[c][j] = (bf16)qb[r * DH + 32 * c + amap(j, half)];
  }

  v8f Oacc[4];
#pragma unroll
  for (int f = 0; f < 4; ++f) Oacc[f] = vzero8();
  float mrow[8], lrow[8];
#pragma unroll
  for (int i = 0; i < 8; ++i) { mrow[i] = -3.0e38f; lrow[i] = 0.f; }

  // prime the TDM pipeline: tile 0 -> buffer 0
  if (wave == 0) tdm_load_tile(Kg, lds_byte_off(&KrowBuf[0][0]));

  for (int tile = 0; tile < NTS; ++tile) {
    __builtin_amdgcn_s_wait_tensorcnt(0);   // only wave 0 has TENSORcnt > 0
    __syncthreads();                        // K tile for 'tile' visible to all
    if (wave == 0 && tile + 1 < NTS)        // kick next DMA while we compute
      tdm_load_tile(Kg + (size_t)(tile + 1) * TT * DH,
                    lds_byte_off(&KrowBuf[(tile + 1) & 1][0]));
    stage_vtr(Vg, tile * TT, Vtr, tid);     // transposed V via VALU path
    if (tile + 1 < NTS)
      __builtin_prefetch(Vg + (size_t)(tile + 1) * TT * DH + tid * 32, 0, 1);
    __syncthreads();
    const bf16* Krow = &KrowBuf[tile & 1][0];

    // S = q K^T for the full 16x128 strip
    v8f s[8];
#pragma unroll
    for (int nb = 0; nb < 8; ++nb) {
      v16bf b0 = bfrag(Krow, DH, 0,  nb * 16, lane);
      v16bf b1 = bfrag(Krow, DH, 32, nb * 16, lane);
      v8f acc = vzero8();
      acc = wmma_bf16(aq[0], b0, acc);
      acc = wmma_bf16(aq[1], b1, acc);
#pragma unroll
      for (int i = 0; i < 8; ++i) acc[i] *= SCALE;
      s[nb] = acc;
    }
    // tile-wide online softmax (one reduction set per row per tile)
#pragma unroll
    for (int i = 0; i < 8; ++i) {
      float rm = s[0][i];
#pragma unroll
      for (int nb = 1; nb < 8; ++nb) rm = fmaxf(rm, s[nb][i]);
      rm = redmax16(rm);
      float mnew  = fmaxf(mrow[i], rm);
      float alpha = __expf(mrow[i] - mnew);
      mrow[i]     = mnew;
      float rs = 0.f;
#pragma unroll
      for (int nb = 0; nb < 8; ++nb) {
        float p = __expf(s[nb][i] - mnew);
        s[nb][i] = p;
        rs += p;
      }
      rs = redsum16(rs);
      lrow[i] = lrow[i] * alpha + rs;
#pragma unroll
      for (int f = 0; f < 4; ++f) Oacc[f][i] *= alpha;
    }
    // P strip D-layout -> LDS -> A-layout, then O += P*V
#pragma unroll
    for (int i = 0; i < 8; ++i) {
      const int row = half * 8 + i;
#pragma unroll
      for (int nb = 0; nb < 8; ++nb)
        myscr[row * 128 + nb * 16 + ncol] = (bf16)s[nb][i];
    }
    asm volatile("s_wait_dscnt 0" ::: "memory");
#pragma unroll
    for (int c = 0; c < 4; ++c) {
      v16bf ap = afrag(myscr, 128, c * 32, lane);
#pragma unroll
      for (int f = 0; f < 4; ++f)
        Oacc[f] = wmma_bf16(ap, bfrag(Vtr, TT, c * 32, f * 16, lane), Oacc[f]);
    }
  }

  // write partial results (unnormalized O, per-row m and l)
  const size_t obase = (size_t)blk * TQ * DH;
#pragma unroll
  for (int f = 0; f < 4; ++f)
#pragma unroll
    for (int i = 0; i < 8; ++i) {
      const int r = r0 + half * 8 + i;
      O_part[obase + (size_t)r * DH + f * 16 + ncol] = Oacc[f][i];
    }
  if (ncol == 0) {
#pragma unroll
    for (int i = 0; i < 8; ++i) {
      const int r = r0 + half * 8 + i;
      m_part[blk * TQ + r] = mrow[i];
      l_part[blk * TQ + r] = lrow[i];
    }
  }
}

// ============ combine partials: O, loss, dO, D, softmax stats ============
__global__ __launch_bounds__(256, 1)
void combine_kernel(const float* __restrict__ m_part,
                    const float* __restrict__ l_part,
                    const float* __restrict__ O_part,
                    float* __restrict__ mrow_g, float* __restrict__ linv_g,
                    float* __restrict__ Drow_g, bf16* __restrict__ dO_bf,
                    float* __restrict__ lossOut /* [BH] */) {
  __shared__ float red[256];
  const int bh = blockIdx.x, tid = threadIdx.x;
  const int r = tid >> 1, hf = tid & 1;   // 128 rows x 2 d-halves

  float mp[SPLIT], w[SPLIT];
  float m = -3.0e38f;
#pragma unroll
  for (int sp = 0; sp < SPLIT; ++sp) {
    mp[sp] = m_part[(bh * SPLIT + sp) * TQ + r];
    m = fmaxf(m, mp[sp]);
  }
  float l = 0.f;
#pragma unroll
  for (int sp = 0; sp < SPLIT; ++sp) {
    float e = __expf(mp[sp] - m);
    l += l_part[(bh * SPLIT + sp) * TQ + r] * e;
    w[sp] = e;
  }
  const float linv = 1.0f / l;
#pragma unroll
  for (int sp = 0; sp < SPLIT; ++sp) w[sp] *= linv;

  float ss = 0.f;
  const int dbase = hf * 32;
  for (int dd = 0; dd < 32; ++dd) {
    const int d = dbase + dd;
    float o = 0.f;
#pragma unroll
    for (int sp = 0; sp < SPLIT; ++sp)
      o += O_part[((size_t)(bh * SPLIT + sp) * TQ + r) * DH + d] * w[sp];
    ss += o * o;
    dO_bf[((size_t)bh * TQ + r) * DH + d] = (bf16)(2.0f * INVN * o);
  }
  const float ssrow = ss + __shfl_xor(ss, 1, 32);
  if (hf == 0) {
    mrow_g[bh * TQ + r] = m;
    linv_g[bh * TQ + r] = linv;
    Drow_g[bh * TQ + r] = 2.0f * INVN * ssrow;   // D = rowsum(O .* dO)
  }
  red[tid] = ss;                                  // deterministic loss reduce
  __syncthreads();
  for (int s2 = 128; s2 > 0; s2 >>= 1) {
    if (tid < s2) red[tid] += red[tid + s2];
    __syncthreads();
  }
  if (tid == 0) lossOut[bh] = red[0];    // raw sum-sq; finalize applies 1/N
}

// ============ backward partial: recompute + gq over one kv split ============
__global__ __launch_bounds__(256, 1)
void bwd_partial_kernel(const float* __restrict__ qbuf,
                        const bf16* __restrict__ Kbf,
                        const bf16* __restrict__ Vbf,
                        const bf16* __restrict__ dO_bf,
                        const float* __restrict__ mrow_g,
                        const float* __restrict__ linv_g,
                        const float* __restrict__ Drow_g,
                        float* __restrict__ gq_part) {
  __shared__ alignas(32) bf16 KrowBuf[2][TT * DH];   // TDM double buffers
  __shared__ alignas(32) bf16 VrowBuf[2][TT * DH];
  __shared__ alignas(32) bf16 Ktr [DH * TT];
  __shared__ alignas(32) bf16 scratch[8 * 16 * 32];  // per-wave gS chunk

  const int blk = blockIdx.x;
  const int bh = blk >> 2, sp = blk & 3;
  const int tid = threadIdx.x;
  const int wave = tid >> 5, lane = tid & 31;
  const int half = lane >> 4, ncol = lane & 15;
  const int r0 = wave * 16;

  const float* qb = qbuf + (size_t)bh * TQ * DH;
  const bf16* Kg  = Kbf + ((size_t)bh * TKV + (size_t)sp * TKS) * DH;
  const bf16* Vg  = Vbf + ((size_t)bh * TKV + (size_t)sp * TKS) * DH;
  const bf16* dOb = dO_bf + (size_t)bh * TQ * DH;
  bf16* myscr     = scratch + wave * (16 * 32);

  v16bf aq[2], adO[2];
  {
    const int r = r0 + (lane & 15);
#pragma unroll
    for (int c = 0; c < 2; ++c)
#pragma unroll
      for (int j = 0; j < 16; ++j) {
        aq[c][j]  = (bf16)qb[r * DH + 32 * c + amap(j, half)];
        adO[c][j] = dOb[r * DH + 32 * c + amap(j, half)];
      }
  }
  float mr[8], li[8], Dr[8];
#pragma unroll
  for (int i = 0; i < 8; ++i) {
    const int r = r0 + half * 8 + i;
    mr[i] = mrow_g[bh * TQ + r];
    li[i] = linv_g[bh * TQ + r];
    Dr[i] = Drow_g[bh * TQ + r];
  }

  v8f gq[4];
#pragma unroll
  for (int f = 0; f < 4; ++f) gq[f] = vzero8();

  if (wave == 0) {                          // prime both DMA streams
    tdm_load_tile(Kg, lds_byte_off(&KrowBuf[0][0]));
    tdm_load_tile(Vg, lds_byte_off(&VrowBuf[0][0]));
  }

  for (int tile = 0; tile < NTS; ++tile) {
    __builtin_amdgcn_s_wait_tensorcnt(0);
    __syncthreads();
    if (wave == 0 && tile + 1 < NTS) {
      tdm_load_tile(Kg + (size_t)(tile + 1) * TT * DH,
                    lds_byte_off(&KrowBuf[(tile + 1) & 1][0]));
      tdm_load_tile(Vg + (size_t)(tile + 1) * TT * DH,
                    lds_byte_off(&VrowBuf[(tile + 1) & 1][0]));
    }
    stage_ktr(Kg, tile * TT, Ktr, tid);     // transposed K via VALU path
    if (tile + 1 < NTS)
      __builtin_prefetch(Kg + (size_t)(tile + 1) * TT * DH + tid * 32, 0, 1);
    __syncthreads();
    const bf16* Krow = &KrowBuf[tile & 1][0];
    const bf16* Vrow = &VrowBuf[tile & 1][0];

#pragma unroll
    for (int c = 0; c < 4; ++c) {
      v8f s0 = vzero8(), s1 = vzero8();   // recompute raw scores
      s0 = wmma_bf16(aq[0], bfrag(Krow, DH, 0,  c * 32,      lane), s0);
      s0 = wmma_bf16(aq[1], bfrag(Krow, DH, 32, c * 32,      lane), s0);
      s1 = wmma_bf16(aq[0], bfrag(Krow, DH, 0,  c * 32 + 16, lane), s1);
      s1 = wmma_bf16(aq[1], bfrag(Krow, DH, 32, c * 32 + 16, lane), s1);
      v8f g0 = vzero8(), g1 = vzero8();   // gA = dO * V^T
      g0 = wmma_bf16(adO[0], bfrag(Vrow, DH, 0,  c * 32,      lane), g0);
      g0 = wmma_bf16(adO[1], bfrag(Vrow, DH, 32, c * 32,      lane), g0);
      g1 = wmma_bf16(adO[0], bfrag(Vrow, DH, 0,  c * 32 + 16, lane), g1);
      g1 = wmma_bf16(adO[1], bfrag(Vrow, DH, 32, c * 32 + 16, lane), g1);
#pragma unroll
      for (int i = 0; i < 8; ++i) {       // gS = P * (gA - D)
        float p0  = __expf(s0[i] * SCALE - mr[i]) * li[i];
        float p1  = __expf(s1[i] * SCALE - mr[i]) * li[i];
        float gs0 = p0 * (g0[i] - Dr[i]);
        float gs1 = p1 * (g1[i] - Dr[i]);
        const int row = half * 8 + i;
        myscr[row * 32 + ncol]      = (bf16)gs0;
        myscr[row * 32 + 16 + ncol] = (bf16)gs1;
      }
      asm volatile("s_wait_dscnt 0" ::: "memory");
      v16bf ags = afrag(myscr, 32, 0, lane);
#pragma unroll
      for (int f = 0; f < 4; ++f)         // gq += gS * K
        gq[f] = wmma_bf16(ags, bfrag(Ktr, TT, c * 32, f * 16, lane), gq[f]);
    }
  }
  const size_t gbase = (size_t)blk * TQ * DH;
#pragma unroll
  for (int f = 0; f < 4; ++f)
#pragma unroll
    for (int i = 0; i < 8; ++i) {
      const int r = r0 + half * 8 + i;
      gq_part[gbase + (size_t)r * DH + f * 16 + ncol] = gq[f][i];
    }
}

// ============ deterministic q update ============
__global__ void q_update_kernel(float* __restrict__ qbuf,
                                const float* __restrict__ gq_part) {
  const int i = blockIdx.x * blockDim.x + threadIdx.x;  // BH*TQ*DH
  const int bh = i >> 13, rd = i & 8191;
  float g = 0.f;
#pragma unroll
  for (int sp = 0; sp < SPLIT; ++sp)
    g += gq_part[(((size_t)(bh * SPLIT + sp)) << 13) + rd];
  qbuf[i] -= LRc * SCALE * g;
}

// ---------- prep / finalize kernels ----------
__global__ void convert_kv_kernel(const float* __restrict__ K,
                                  const float* __restrict__ V,
                                  bf16* __restrict__ Kb, bf16* __restrict__ Vb,
                                  long n4) {
  long i = (long)blockIdx.x * blockDim.x + threadIdx.x;
  if (i >= n4) return;
  float4 k = ((const float4*)K)[i];
  float4 v = ((const float4*)V)[i];
  v4bf kb = { (bf16)k.x, (bf16)k.y, (bf16)k.z, (bf16)k.w };
  v4bf vb = { (bf16)v.x, (bf16)v.y, (bf16)v.z, (bf16)v.w };
  *(v4bf*)(Kb + i * 4) = kb;
  *(v4bf*)(Vb + i * 4) = vb;
}

__global__ void q_init_kernel(const float* __restrict__ queries,
                              float* __restrict__ qbuf) {
  int i  = blockIdx.x * blockDim.x + threadIdx.x;
  int bh = i >> 13;
  int t  = (i >> 6) & 127;
  int d  = i & 63;
  int b  = bh >> 5, h = bh & 31;
  qbuf[i] = queries[(size_t)(b * TQ + t) * Dd + h * DH + d];
}

__global__ void finalize_kernel(const float* __restrict__ qbuf,
                                const float* __restrict__ lossPartial,
                                float* __restrict__ out) {
  int i = blockIdx.x * blockDim.x + threadIdx.x;
  if (i < BH * TQ * DH) {
    int bh = i >> 13;
    int t  = (i >> 6) & 127;
    int d  = i & 63;
    int b  = bh >> 5, h = bh & 31;
    out[(size_t)(b * TQ + t) * Dd + h * DH + d] = qbuf[i];
  }
  if (blockIdx.x == 0 && threadIdx.x < NSTEPS) {
    float s = 0.f;
    for (int j = 0; j < BH; ++j) s += lossPartial[threadIdx.x * BH + j];
    out[(size_t)BH * TQ * DH + threadIdx.x] = s * INVN;
  }
}

// ---------- entry ----------
extern "C" void kernel_launch(void* const* d_in, const int* in_sizes, int n_in,
                              void* d_out, int out_size, void* d_ws, size_t ws_size,
                              hipStream_t stream) {
  const float* queries = (const float*)d_in[0];
  const float* keys    = (const float*)d_in[1];
  const float* values  = (const float*)d_in[2];
  (void)in_sizes; (void)n_in; (void)out_size; (void)ws_size;
  // num_steps is a setup constant (16); a device scalar cannot be read during
  // graph capture, so the step count is compiled in.

  const size_t kvElems = (size_t)BH * TKV * DH;  // 33,554,432
  char* p = (char*)d_ws;
  bf16*  Kb    = (bf16*)p;  p += kvElems * sizeof(bf16);                 // 64 MB
  bf16*  Vb    = (bf16*)p;  p += kvElems * sizeof(bf16);                 // 64 MB
  float* qbuf  = (float*)p; p += (size_t)BH * TQ * DH * sizeof(float);   // 2 MB
  float* m_part = (float*)p; p += (size_t)BH * SPLIT * TQ * sizeof(float);
  float* l_part = (float*)p; p += (size_t)BH * SPLIT * TQ * sizeof(float);
  float* O_part = (float*)p; p += (size_t)BH * SPLIT * TQ * DH * sizeof(float); // 8 MB
  float* mrow_g = (float*)p; p += (size_t)BH * TQ * sizeof(float);
  float* linv_g = (float*)p; p += (size_t)BH * TQ * sizeof(float);
  float* Drow_g = (float*)p; p += (size_t)BH * TQ * sizeof(float);
  bf16*  dO_bf  = (bf16*)p;  p += (size_t)BH * TQ * DH * sizeof(bf16);   // 1 MB
  float* gq_part = (float*)p; p += (size_t)BH * SPLIT * TQ * DH * sizeof(float); // 8 MB
  float* lossPartial = (float*)p;                                        // 16*64 f32

  {
    long n4 = (long)(kvElems / 4);
    convert_kv_kernel<<<(int)((n4 + 255) / 256), 256, 0, stream>>>(keys, values, Kb, Vb, n4);
  }
  q_init_kernel<<<(BH * TQ * DH) / 256, 256, 0, stream>>>(queries, qbuf);

  for (int s = 0; s < NSTEPS; ++s) {
    fwd_partial_kernel<<<BH * SPLIT, 256, 0, stream>>>(qbuf, Kb, Vb,
                                                       m_part, l_part, O_part);
    combine_kernel<<<BH, 256, 0, stream>>>(m_part, l_part, O_part,
                                           mrow_g, linv_g, Drow_g, dO_bf,
                                           lossPartial + s * BH);
    bwd_partial_kernel<<<BH * SPLIT, 256, 0, stream>>>(qbuf, Kb, Vb, dO_bf,
                                                       mrow_g, linv_g, Drow_g,
                                                       gq_part);
    q_update_kernel<<<(BH * TQ * DH) / 256, 256, 0, stream>>>(qbuf, gq_part);
  }

  finalize_kernel<<<(BH * TQ * DH) / 256, 256, 0, stream>>>(qbuf, lossPartial, (float*)d_out);
}
